// LSTMCell_49160195670661
// MI455X (gfx1250) — matrix operand order repaired and checked by reading, not compiled, not run on hardware
//
#include <hip/hip_runtime.h>
#include <hip/hip_bf16.h>
#include <math.h>

// Problem sizes (fixed by the reference).
#define B_SZ  8192
#define IN_SZ 2048
#define H_SZ  2048
#define K_SZ  (IN_SZ + H_SZ)   // 4096

typedef __attribute__((ext_vector_type(16))) __bf16 v16bf;
typedef __attribute__((ext_vector_type(8)))  __bf16 v8bf;
typedef __attribute__((ext_vector_type(8)))  float  v8f;
typedef __attribute__((ext_vector_type(4)))  int    i32x4;

// LDS row padding: 64 halfs data + 8 halfs pad = 72 (144 B) -> conflict-free strides
#define LDS_ROW 72

#define AS1 __attribute__((address_space(1)))
#define AS3 __attribute__((address_space(3)))

// gfx1250 async global->LDS staging (ASYNCcnt path), with sync fallback.
#if defined(__has_builtin)
#  if __has_builtin(__builtin_amdgcn_global_load_async_to_lds_b128)
#    define USE_ASYNC_LDS 1
#  endif
#endif

__device__ __forceinline__ void wait_asynccnt0() {
#if defined(__has_builtin)
#  if __has_builtin(__builtin_amdgcn_s_wait_asynccnt)
    __builtin_amdgcn_s_wait_asynccnt(0);
    return;
#  endif
#endif
    asm volatile("s_wait_asynccnt 0x0" ::: "memory");
}

// ---------- helpers ----------
__device__ __forceinline__ unsigned short f2bf_rne(float x) {
    unsigned u = __float_as_uint(x);
    unsigned r = u + 0x7FFFu + ((u >> 16) & 1u);
    return (unsigned short)(r >> 16);
}

__device__ __forceinline__ float sigmoid_f(float x) {
    return 1.0f / (1.0f + __expf(-x));
}

__device__ __forceinline__ float tanh_f(float x) {
    return __builtin_fmaf(2.0f, sigmoid_f(2.0f * x), -1.0f);
}

// ---------- phase 1a: bf16 combined = [input | h_prev], [B][K] ----------
__global__ void __launch_bounds__(256)
convert_combined(const float* __restrict__ inp, const float* __restrict__ hprev,
                 unsigned short* __restrict__ Cb) {
    size_t t    = (size_t)blockIdx.x * blockDim.x + threadIdx.x;
    size_t base = t * 4;
    int row = (int)(base / K_SZ);
    int col = (int)(base % K_SZ);
    const float* src = (col < IN_SZ)
        ? (inp   + (size_t)row * IN_SZ + col)
        : (hprev + (size_t)row * H_SZ  + (col - IN_SZ));
    float4 v = *(const float4*)src;
    ushort4 o;
    o.x = f2bf_rne(v.x); o.y = f2bf_rne(v.y);
    o.z = f2bf_rne(v.z); o.w = f2bf_rne(v.w);
    *(ushort4*)(Cb + base) = o;
}

// ---------- phase 1b: bf16 W_all, gate-major [4][H][K] ----------
__global__ void __launch_bounds__(256)
convert_weights(const float* __restrict__ Wf, const float* __restrict__ Wi,
                const float* __restrict__ Wg, const float* __restrict__ Wo,
                unsigned short* __restrict__ Wb) {
    size_t t    = (size_t)blockIdx.x * blockDim.x + threadIdx.x;
    size_t base = t * 4;
    const size_t gsz = (size_t)H_SZ * K_SZ;
    int    gate = (int)(base / gsz);
    size_t rem  = base % gsz;
    const float* src = (gate == 0) ? Wf : (gate == 1) ? Wi : (gate == 2) ? Wg : Wo;
    float4 v = *(const float4*)(src + rem);
    ushort4 o;
    o.x = f2bf_rne(v.x); o.y = f2bf_rne(v.y);
    o.z = f2bf_rne(v.z); o.w = f2bf_rne(v.w);
    *(ushort4*)(Wb + base) = o;
}

// ---------- phase 2: fused 4-gate WMMA GEMM + LSTM cell update ----------
// Block = 256 threads = 8 waves (4 along M x 2 along N), Mrep = 2 per wave.
// Block tile 128(M) x 32(N) per gate; k-step 64 per LDS stage; 16 WMMAs/stage/wave.
__global__ void __launch_bounds__(256)
lstm_wmma(const unsigned short* __restrict__ Cb,   // [B][K]    bf16 combined
          const unsigned short* __restrict__ Wb,   // [4][H][K] bf16 weights
          const float* __restrict__ bf, const float* __restrict__ bi,
          const float* __restrict__ bg, const float* __restrict__ bo,
          const float* __restrict__ c_prev,
          float* __restrict__ h_out,
          float* __restrict__ c_out) {

    __shared__ unsigned short sA[128 * LDS_ROW];      // 18 KB
    __shared__ unsigned short sB[4 * 32 * LDS_ROW];   // 18 KB

    const int tid   = threadIdx.x;
    const int lane  = tid & 31;
    const int wave  = tid >> 5;
    const int wm    = wave & 3;
    const int wn    = wave >> 2;
    const int l15   = lane & 15;
    const int lhalf = lane >> 4;

    const int m0 = blockIdx.x * 128;
    const int n0 = blockIdx.y * 32;

    // ---- staging maps; explicit RUNNING pointers (bumped +64 halfs per stage) ----
    const int srow = tid >> 3;            // 0..31
    const int soff = (tid & 7) * 8;       // halfs
    const size_t gsz = (size_t)H_SZ * K_SZ;

    const unsigned short* pA[4];
    int lA[4];
    #pragma unroll
    for (int i = 0; i < 4; ++i) {
        const int row = srow + i * 32;
        pA[i] = Cb + (size_t)(m0 + row) * K_SZ + soff;
        lA[i] = row * LDS_ROW + soff;
    }
    const unsigned short* pB[4];
    int lB[4];
    #pragma unroll
    for (int g = 0; g < 4; ++g) {
        pB[g] = Wb + (size_t)g * gsz + (size_t)(n0 + srow) * K_SZ + soff;
        lB[g] = (g * 32 + srow) * LDS_ROW + soff;
    }

    // ---- fragment read offsets (halfs) ----
    int aBase[2];
    #pragma unroll
    for (int r = 0; r < 2; ++r)
        aBase[r] = ((wm * 2 + r) * 16 + l15) * LDS_ROW + lhalf * 8;
    int bBase[4];
    #pragma unroll
    for (int g = 0; g < 4; ++g)
        bBase[g] = (g * 32 + l15) * LDS_ROW + lhalf * 8;

    v8f acc[4][2];
    #pragma unroll
    for (int g = 0; g < 4; ++g) {
        acc[g][0] = (v8f){};
        acc[g][1] = (v8f){};
    }

    for (int ks = 0; ks < K_SZ / 64; ++ks) {
#if defined(USE_ASYNC_LDS)
        // fire-and-forget: per-lane 16B global->LDS, tracked by ASYNCcnt.
        // Param 1 is AS1 (global) int4*, param 2 the LDS destination.
        #pragma unroll
        for (int i = 0; i < 4; ++i)
            __builtin_amdgcn_global_load_async_to_lds_b128(
                (AS1 i32x4*)pA[i], (AS3 i32x4*)&sA[lA[i]], 0, 0);
        #pragma unroll
        for (int g = 0; g < 4; ++g)
            __builtin_amdgcn_global_load_async_to_lds_b128(
                (AS1 i32x4*)pB[g], (AS3 i32x4*)&sB[lB[g]], 0, 0);
#else
        #pragma unroll
        for (int i = 0; i < 4; ++i)
            *(uint4*)&sA[lA[i]] = *(const uint4*)pA[i];
        #pragma unroll
        for (int g = 0; g < 4; ++g)
            *(uint4*)&sB[lB[g]] = *(const uint4*)pB[g];
#endif
        // prefetch next stage (global_prefetch_b8) + bump running pointers
        if (ks + 1 < K_SZ / 64) {
            #pragma unroll
            for (int i = 0; i < 4; ++i) __builtin_prefetch(pA[i] + 64, 0, 1);
            #pragma unroll
            for (int g = 0; g < 4; ++g) __builtin_prefetch(pB[g] + 64, 0, 1);
        }
        #pragma unroll
        for (int i = 0; i < 4; ++i) pA[i] += 64;
        #pragma unroll
        for (int g = 0; g < 4; ++g) pB[g] += 64;

#if defined(USE_ASYNC_LDS)
        wait_asynccnt0();
#endif
        __syncthreads();

        #pragma unroll
        for (int kc = 0; kc < 2; ++kc) {
            // A fragments (ISA 16-bit A layout: lane-half selects K-halves)
            v16bf afrag[2];
            #pragma unroll
            for (int r = 0; r < 2; ++r) {
                v8bf alo = *(const v8bf*)&sA[aBase[r] + kc * 32];
                v8bf ahi = *(const v8bf*)&sA[aBase[r] + kc * 32 + 16];
                afrag[r] = __builtin_shufflevector(alo, ahi,
                            0, 1, 2, 3, 4, 5, 6, 7, 8, 9, 10, 11, 12, 13, 14, 15);
            }
            // all 4 gate B fragments live at once -> WMMAs not gated on dscnt==0
            v16bf bfrag[4];
            #pragma unroll
            for (int g = 0; g < 4; ++g) {
                v8bf blo = *(const v8bf*)&sB[bBase[g] + kc * 32];
                v8bf bhi = *(const v8bf*)&sB[bBase[g] + kc * 32 + 16];
                bfrag[g] = __builtin_shufflevector(blo, bhi,
                            0, 1, 2, 3, 4, 5, 6, 7, 8, 9, 10, 11, 12, 13, 14, 15);
            }
            #pragma unroll
            for (int g = 0; g < 4; ++g) {
                acc[g][0] = __builtin_amdgcn_wmma_f32_16x16x32_bf16(
                                false, afrag[0], false, bfrag[g],
                                (short)0, acc[g][0], false, false);
                acc[g][1] = __builtin_amdgcn_wmma_f32_16x16x32_bf16(
                                false, afrag[1], false, bfrag[g],
                                (short)0, acc[g][1], false, false);
            }
        }
        __syncthreads();
    }

    // ---- epilogue: bias + gate nonlinearities + cell update ----
    const int ncol  = n0 + wn * 16 + l15;
    const float biasf = bf[ncol];
    const float biasi = bi[ncol];
    const float biasg = bg[ncol];
    const float biaso = bo[ncol];

    #pragma unroll
    for (int rr = 0; rr < 2; ++rr) {
        const int mbase = m0 + (wm * 2 + rr) * 16 + lhalf * 8;
        #pragma unroll
        for (int r = 0; r < 8; ++r) {
            const int m = mbase + r;
            const size_t off = (size_t)m * H_SZ + ncol;
            float fg = sigmoid_f(acc[0][rr][r] + biasf);
            float ig = sigmoid_f(acc[1][rr][r] + biasi);
            float gg = tanh_f   (acc[2][rr][r] + biasg);
            float og = sigmoid_f(acc[3][rr][r] + biaso);
            float c  = fg * c_prev[off] + ig * gg;
            float h  = og * tanh_f(c);
            h_out[off] = h;
            c_out[off] = c;
        }
    }
}

// ---------- host entry ----------
extern "C" void kernel_launch(void* const* d_in, const int* in_sizes, int n_in,
                              void* d_out, int out_size, void* d_ws, size_t ws_size,
                              hipStream_t stream) {
    (void)in_sizes; (void)n_in; (void)out_size; (void)ws_size;

    const float* input  = (const float*)d_in[0];
    const float* h_prev = (const float*)d_in[1];
    const float* c_prev = (const float*)d_in[2];
    const float* Wf     = (const float*)d_in[3];
    const float* bf     = (const float*)d_in[4];
    const float* Wi     = (const float*)d_in[5];
    const float* bi     = (const float*)d_in[6];
    const float* Wg     = (const float*)d_in[7];
    const float* bg     = (const float*)d_in[8];
    const float* Wo     = (const float*)d_in[9];
    const float* bo     = (const float*)d_in[10];

    unsigned short* Cb = (unsigned short*)d_ws;                 // 64 MiB
    unsigned short* Wb = Cb + (size_t)B_SZ * K_SZ;              // 64 MiB

    float* h_out = (float*)d_out;
    float* c_out = h_out + (size_t)B_SZ * H_SZ;

    {
        const size_t nthreads = (size_t)B_SZ * K_SZ / 4;
        convert_combined<<<dim3((unsigned)(nthreads / 256)), dim3(256), 0, stream>>>(
            input, h_prev, Cb);
    }
    {
        const size_t nthreads = (size_t)4 * H_SZ * K_SZ / 4;
        convert_weights<<<dim3((unsigned)(nthreads / 256)), dim3(256), 0, stream>>>(
            Wf, Wi, Wg, Wo, Wb);
    }
    lstm_wmma<<<dim3(B_SZ / 128, H_SZ / 32), dim3(256), 0, stream>>>(
        Cb, Wb, bf, bi, bg, bo, c_prev, h_out, c_out);
}
